// AKT_4346506903942
// MI455X (gfx1250) — matrix operand order repaired and checked by reading, not compiled, not run on hardware
//
#include <hip/hip_runtime.h>
#include <hip/hip_bf16.h>
#include <math.h>

// ---------------------------------------------------------------------------
// Fused decay-attention for MI455X (gfx1250, wave32, WMMA).
//   B=4, S=2048, E=256, H=8, DK=32, tokens NT = B*S = 8192.
// Strategy: keep the 16 x 2048 score row-panel resident in LDS (320KB/WGP),
// never touch HBM with [B,H,S,S] tensors; all GEMMs via v_wmma_f32_16x16x32_bf16.
// Attention block uses 16 waves so the VALU-bound scan stage gets one wave
// per query row and 4 waves/SIMD of latency hiding.
// ---------------------------------------------------------------------------

typedef __attribute__((ext_vector_type(16))) __bf16 v16bf;
typedef __attribute__((ext_vector_type(8)))  __bf16 v8bf;
typedef __attribute__((ext_vector_type(8)))  float  v8f;

#define WMMA_BF16(a, b, c) \
  __builtin_amdgcn_wmma_f32_16x16x32_bf16(false, (a), false, (b), (short)0, (c), false, false)

static __device__ inline v16bf cat8(v8bf lo, v8bf hi) {
  return __builtin_shufflevector(lo, hi, 0, 1, 2, 3, 4, 5, 6, 7,
                                         8, 9, 10, 11, 12, 13, 14, 15);
}

// ---------------------------------------------------------------------------
// fp32 -> bf16 conversion
// ---------------------------------------------------------------------------
__global__ void AKT_cvt_bf16_kernel(const float* __restrict__ in,
                                    __bf16* __restrict__ out, int n) {
  int i = blockIdx.x * blockDim.x + threadIdx.x;
  if (i < n) out[i] = (__bf16)in[i];
}

// ---------------------------------------------------------------------------
// QKV projection: out = X @ W^T + b, reshaped to heads.
//   X: [NT, 256] bf16 row-major. W: [256, 256] bf16 row-major (row = out chan).
//   mode 0: Q -> qh[bh][s][dk], scaled by 1/sqrt(32)
//   mode 1: K -> kh[bh][s][dk]
//   mode 2: V -> vT[bh][dk][s]   (transposed for contiguous B-fragments)
// Block = 128 threads (4 waves); wave computes one 16x16 output tile.
// ---------------------------------------------------------------------------
__global__ void AKT_proj_qkv_kernel(const __bf16* __restrict__ X,
                                    const __bf16* __restrict__ W,
                                    const float* __restrict__ bias,
                                    __bf16* __restrict__ out, int mode) {
  const int lane = threadIdx.x & 31;
  const int wave = threadIdx.x >> 5;
  const int row0 = blockIdx.y * 16;
  const int col0 = blockIdx.x * 64 + wave * 16;

  const int m   = lane & 15;
  const int ka  = (lane >> 4) * 8;    // unused: A-frag also uses kb2 (16 contiguous)
  const int kb2 = (lane >> 4) * 16;   // B-frag K base (16 contiguous)
  const int n   = col0 + m;

  v8f acc = {0.f, 0.f, 0.f, 0.f, 0.f, 0.f, 0.f, 0.f};
  for (int kb = 0; kb < 256; kb += 32) {
    const __bf16* ap = X + (row0 + m) * 256 + kb + kb2;
    v16bf afr = cat8(*(const v8bf*)ap, *(const v8bf*)(ap + 8));
    // B[k][n] = W[n][k] -> 16 contiguous bf16 from row n of W
    const __bf16* bp = W + n * 256 + kb + kb2;
    v16bf bfr = cat8(*(const v8bf*)bp, *(const v8bf*)(bp + 8));
    acc = WMMA_BF16(afr, bfr, acc);
  }

  const float bn = bias[n];
  const int Mo = (lane >> 4) * 8;
#pragma unroll
  for (int v = 0; v < 8; ++v) {
    int t = row0 + Mo + v;            // global token
    int c = col0 + m;                 // output channel
    float val = acc[v] + bn;
    int b = t >> 11, s = t & 2047;
    int h = c >> 5, d = c & 31;
    if (mode == 0) {
      val *= 0.17677669529663687f;    // 1/sqrt(32) folded into Q
      out[(((b * 8 + h) * 2048) + s) * 32 + d] = (__bf16)val;
    } else if (mode == 1) {
      out[(((b * 8 + h) * 2048) + s) * 32 + d] = (__bf16)val;
    } else {
      out[(((b * 8 + h) * 32) + d) * 2048 + s] = (__bf16)val;
    }
  }
}

// ---------------------------------------------------------------------------
// Output projection: out_f32 = att @ Wo^T + bo
// ---------------------------------------------------------------------------
__global__ void AKT_proj_out_kernel(const __bf16* __restrict__ X,
                                    const __bf16* __restrict__ W,
                                    const float* __restrict__ bias,
                                    float* __restrict__ out) {
  const int lane = threadIdx.x & 31;
  const int wave = threadIdx.x >> 5;
  const int row0 = blockIdx.y * 16;
  const int col0 = blockIdx.x * 64 + wave * 16;

  const int m   = lane & 15;
  const int ka  = (lane >> 4) * 8;
  const int kb2 = (lane >> 4) * 16;
  const int n   = col0 + m;

  v8f acc = {0.f, 0.f, 0.f, 0.f, 0.f, 0.f, 0.f, 0.f};
  for (int kb = 0; kb < 256; kb += 32) {
    const __bf16* ap = X + (row0 + m) * 256 + kb + kb2;
    v16bf afr = cat8(*(const v8bf*)ap, *(const v8bf*)(ap + 8));
    const __bf16* bp = W + n * 256 + kb + kb2;
    v16bf bfr = cat8(*(const v8bf*)bp, *(const v8bf*)(bp + 8));
    acc = WMMA_BF16(afr, bfr, acc);
  }
  const float bn = bias[n];
  const int Mo = (lane >> 4) * 8;
#pragma unroll
  for (int v = 0; v < 8; ++v) {
    int t = row0 + Mo + v;
    int c = col0 + m;
    out[t * 256 + c] = acc[v] + bn;
  }
}

// ---------------------------------------------------------------------------
// Fused decay attention. One workgroup = (b,h, 16-query row tile).
// Block = 512 threads = 16 waves (one wave per query row in the scan stage).
// Dynamic LDS:
//   sc  : [16][2048] fp32 score panel        (128 KB)
//   pb  : [16][2048] bf16 prob panel         ( 64 KB)
//   red : [16][16][32] fp32 partial out      ( 32 KB)
// ---------------------------------------------------------------------------
__global__ void AKT_attn_kernel(const __bf16* __restrict__ qh,
                                const __bf16* __restrict__ kh,
                                const __bf16* __restrict__ vT,
                                const float* __restrict__ gammas,
                                __bf16* __restrict__ att) {
  extern __shared__ char smem[];
  float*  sc  = (float*)smem;                                  // 16*2048 f32
  __bf16* pb  = (__bf16*)(smem + 16 * 2048 * 4);               // 16*2048 bf16
  float*  red = (float*)(smem + 16 * 2048 * 4 + 16 * 2048 * 2);// 16*16*32 f32

  const int lane = threadIdx.x & 31;
  const int wave = threadIdx.x >> 5;    // 0..15
  const int rt   = blockIdx.x;          // query row tile (16 rows)
  const int bh   = blockIdx.y;          // b*H + h
  const int b    = bh >> 3, h = bh & 7;

  const __bf16* Q  = qh + (size_t)bh * 2048 * 32;
  const __bf16* K  = kh + (size_t)bh * 2048 * 32;
  const __bf16* Vt = vT + (size_t)bh * 32 * 2048;

  const float gamma = -log1pf(__expf(gammas[h]));   // -softplus

  const int m   = lane & 15;
  const int ka  = (lane >> 4) * 8;
  const int kb2 = (lane >> 4) * 16;
  const int Mo  = (lane >> 4) * 8;

  // ---- Stage 1: scores panel = (Q/sqrt(dk)) @ K^T into LDS --------------
  {
    const __bf16* qp = Q + (rt * 16 + m) * 32 + kb2;
    v16bf afr = cat8(*(const v8bf*)qp, *(const v8bf*)(qp + 8));
    for (int t = wave; t <= rt; t += 16) {          // causal: tiles 0..rt
      if (t + 16 <= rt) {                           // prefetch next K tile
        __builtin_prefetch(K + ((t + 16) * 16 + m) * 32 + kb2, 0, 1);
      }
      const __bf16* kp = K + (t * 16 + m) * 32 + kb2;
      v16bf bfr = cat8(*(const v8bf*)kp, *(const v8bf*)(kp + 8));
      v8f c = {0.f, 0.f, 0.f, 0.f, 0.f, 0.f, 0.f, 0.f};
      c = WMMA_BF16(afr, bfr, c);
#pragma unroll
      for (int v = 0; v < 8; ++v)
        sc[(Mo + v) * 2048 + t * 16 + m] = c[v];
    }
  }
  __syncthreads();

  const int bound = (rt + 1) * 16;        // exclusive key bound for this tile
  const int nch   = (bound + 31) >> 5;    // 32-wide chunks

  // ---- Stages 2+3: per-row softmax, suffix-sum decay, 2nd softmax -------
  {
    const int row = wave;                 // one wave per query row
    const int iq  = rt * 16 + row;        // valid keys: j <= iq
    float* srow = sc + row * 2048;

    // online max/sum of exp (first softmax stats); disttotal == 1 exactly
    float m1 = -1e30f, l1 = 0.f;
    for (int c0 = 0; c0 < nch; ++c0) {
      int j = c0 * 32 + lane;
      if (j <= iq) {
        float sv = srow[j];
        float mn = fmaxf(m1, sv);
        l1 = l1 * __expf(m1 - mn) + __expf(sv - mn);
        m1 = mn;
      }
    }
#pragma unroll
    for (int off = 16; off; off >>= 1) {
      float mo = __shfl_xor(m1, off, 32);
      float lo = __shfl_xor(l1, off, 32);
      float mn = fmaxf(m1, mo);
      l1 = l1 * __expf(m1 - mn) + lo * __expf(mo - mn);
      m1 = mn;
    }
    const float invl1 = 1.0f / l1;

    // sequential chunked inclusive scan: suffix = 1 - cumsum(s1);
    // sc2 = score * clamp(exp(sqrt(suffix*|i-j|)*gamma)); online stats m2/l2.
    float carry = 0.f, m2 = -1e30f, l2 = 0.f;
    for (int c0 = 0; c0 < nch; ++c0) {
      int j = c0 * 32 + lane;
      bool valid = (j <= iq);
      float sv = valid ? srow[j] : 0.f;
      float p1 = valid ? __expf(sv - m1) * invl1 : 0.f;
      float ps = p1;
#pragma unroll
      for (int d = 1; d < 32; d <<= 1) {
        float tu = __shfl_up(ps, d, 32);
        if (lane >= d) ps += tu;
      }
      float tot  = __shfl(ps, 31, 32);
      float cumi = carry + ps;            // inclusive cumsum through j
      carry += tot;
      float sc2;
      if (valid) {
        float suffix = fmaxf(1.0f - cumi, 0.0f);
        float dist = sqrtf(suffix * (float)(iq - j));
        float eff  = fminf(fmaxf(__expf(dist * gamma), 1e-5f), 1e5f);
        sc2 = sv * eff;
        float mn = fmaxf(m2, sc2);
        l2 = l2 * __expf(m2 - mn) + __expf(sc2 - mn);
        m2 = mn;
      } else {
        sc2 = -1e30f;                     // masked
      }
      srow[j] = sc2;                      // overwrite panel in place
    }
#pragma unroll
    for (int off = 16; off; off >>= 1) {
      float mo = __shfl_xor(m2, off, 32);
      float lo = __shfl_xor(l2, off, 32);
      float mn = fmaxf(m2, mo);
      l2 = l2 * __expf(m2 - mn) + lo * __expf(mo - mn);
      m2 = mn;
    }
    const float invl2 = 1.0f / l2;

    __bf16* prow = pb + row * 2048;
    for (int c0 = 0; c0 < nch; ++c0) {
      int j = c0 * 32 + lane;
      float p = (j <= iq) ? __expf(srow[j] - m2) * invl2 : 0.f;
      prow[j] = (__bf16)p;
    }
  }
  __syncthreads();

  // ---- Stage 4: out(16x32) = P(16 x bound) @ V^T; cross-wave reduce -----
  v8f a0 = {0.f, 0.f, 0.f, 0.f, 0.f, 0.f, 0.f, 0.f};
  v8f a1 = {0.f, 0.f, 0.f, 0.f, 0.f, 0.f, 0.f, 0.f};
  for (int c0 = wave; c0 * 32 < bound; c0 += 16) {
    int jb = c0 * 32;
    if (jb + 512 < bound) {                          // prefetch next V chunk
      __builtin_prefetch(Vt + m * 2048 + jb + 512 + kb2, 0, 1);
      __builtin_prefetch(Vt + (16 + m) * 2048 + jb + 512 + kb2, 0, 1);
    }
    const __bf16* pr = pb + m * 2048 + jb + kb2;     // A-frag from prob panel
    v16bf pa = cat8(*(const v8bf*)pr, *(const v8bf*)(pr + 8));
    const __bf16* v0p = Vt + m * 2048 + jb + kb2;    // B-frag, d = m
    v16bf vb0 = cat8(*(const v8bf*)v0p, *(const v8bf*)(v0p + 8));
    const __bf16* v1p = Vt + (16 + m) * 2048 + jb + kb2; // B-frag, d = 16+m
    v16bf vb1 = cat8(*(const v8bf*)v1p, *(const v8bf*)(v1p + 8));
    a0 = WMMA_BF16(pa, vb0, a0);
    a1 = WMMA_BF16(pa, vb1, a1);
  }
  float* rw = red + wave * 512;
#pragma unroll
  for (int v = 0; v < 8; ++v) {
    rw[(Mo + v) * 32 + m]      = a0[v];
    rw[(Mo + v) * 32 + 16 + m] = a1[v];
  }
  __syncthreads();
  {
    int e = threadIdx.x;                  // 512 threads, 512 outputs
    int mm = e >> 5, dd = e & 31;
    float sum = 0.f;
#pragma unroll
    for (int w = 0; w < 16; ++w) sum += red[w * 512 + mm * 32 + dd];
    int s_tok = rt * 16 + mm;
    att[((size_t)(b * 2048 + s_tok)) * 256 + h * 32 + dd] = (__bf16)sum;
  }
}

// ---------------------------------------------------------------------------
// Host launcher
// ---------------------------------------------------------------------------
extern "C" void kernel_launch(void* const* d_in, const int* in_sizes, int n_in,
                              void* d_out, int out_size, void* d_ws, size_t ws_size,
                              hipStream_t stream) {
  (void)in_sizes; (void)n_in; (void)out_size; (void)ws_size;
  const float* q      = (const float*)d_in[0];
  const float* k      = (const float*)d_in[1];
  const float* v      = (const float*)d_in[2];
  // d_in[3] = mask (causal; structure known, not read)
  const float* Wq     = (const float*)d_in[4];
  const float* bq     = (const float*)d_in[5];
  const float* Wk     = (const float*)d_in[6];
  const float* bk     = (const float*)d_in[7];
  const float* Wv     = (const float*)d_in[8];
  const float* bv     = (const float*)d_in[9];
  const float* Wo     = (const float*)d_in[10];
  const float* bo     = (const float*)d_in[11];
  const float* gammas = (const float*)d_in[12];
  float* out = (float*)d_out;

  const size_t NX = (size_t)8192 * 256;   // tokens * E elements
  const size_t NW = (size_t)256 * 256;

  char* ws = (char*)d_ws;
  size_t o = 0;
  __bf16* xq  = (__bf16*)(ws + o); o += NX * 2;
  __bf16* xk  = (__bf16*)(ws + o); o += NX * 2;
  __bf16* xv  = (__bf16*)(ws + o); o += NX * 2;
  __bf16* wqb = (__bf16*)(ws + o); o += NW * 2;
  __bf16* wkb = (__bf16*)(ws + o); o += NW * 2;
  __bf16* wvb = (__bf16*)(ws + o); o += NW * 2;
  __bf16* wob = (__bf16*)(ws + o); o += NW * 2;
  __bf16* qhb = (__bf16*)(ws + o); o += NX * 2;   // [B,H,S,DK]
  __bf16* khb = (__bf16*)(ws + o); o += NX * 2;   // [B,H,S,DK]
  __bf16* vTb = (__bf16*)(ws + o); o += NX * 2;   // [B,H,DK,S]
  __bf16* att = (__bf16*)(ws + o); o += NX * 2;   // [B,S,E]

  // fp32 -> bf16 conversions
  AKT_cvt_bf16_kernel<<<(int)((NX + 255) / 256), 256, 0, stream>>>(q, xq, (int)NX);
  AKT_cvt_bf16_kernel<<<(int)((NX + 255) / 256), 256, 0, stream>>>(k, xk, (int)NX);
  AKT_cvt_bf16_kernel<<<(int)((NX + 255) / 256), 256, 0, stream>>>(v, xv, (int)NX);
  AKT_cvt_bf16_kernel<<<(int)((NW + 255) / 256), 256, 0, stream>>>(Wq, wqb, (int)NW);
  AKT_cvt_bf16_kernel<<<(int)((NW + 255) / 256), 256, 0, stream>>>(Wk, wkb, (int)NW);
  AKT_cvt_bf16_kernel<<<(int)((NW + 255) / 256), 256, 0, stream>>>(Wv, wvb, (int)NW);
  AKT_cvt_bf16_kernel<<<(int)((NW + 255) / 256), 256, 0, stream>>>(Wo, wob, (int)NW);

  // projections: grid (E/64 col groups, NT/16 row tiles), 4 waves per WG
  dim3 pg(4, 512);
  AKT_proj_qkv_kernel<<<pg, 128, 0, stream>>>(xq, wqb, bq, qhb, 0);
  AKT_proj_qkv_kernel<<<pg, 128, 0, stream>>>(xk, wkb, bk, khb, 1);
  AKT_proj_qkv_kernel<<<pg, 128, 0, stream>>>(xv, wvb, bv, vTb, 2);

  // fused attention: grid (S/16 row tiles, B*H), 16 waves, 224 KB dynamic LDS
  const size_t lds = 16 * 2048 * 4 + 16 * 2048 * 2 + 16 * 512 * 4; // 229376 B
  dim3 ag(128, 32);
  AKT_attn_kernel<<<ag, 512, lds, stream>>>(qhb, khb, vTb, gammas, att);

  // output projection (fp32 result)
  AKT_proj_out_kernel<<<pg, 128, 0, stream>>>(att, wob, bo, out);
}
